// HierarchicalDecouplingModule_1262720385566
// MI455X (gfx1250) — compile-verified, matrix operand order checked by reading, not run on hardware
//
#include <hip/hip_runtime.h>

// HierarchicalDecouplingModule for MI455X (gfx1250, wave32, WMMA bf16).
//
// Restructured math:
//   qh[i]   = Wq_i @ tokens[i] + bq_i                       (tiny)
//   r[i,h]  = HD^-0.5 * qh[i,h] @ Wk_i[h-rows]              (tiny, bf16)
//   scores  = X_l @ r^T          -> softmax over l          (thin WMMA GEMM)
//   O       = (sum_l a_l (.) X_l) @ Wv^T                    (WMMA, a folded into A-frags)
//   out     = O @ Wo^T + (Wo@bv + b_out)                    (WMMA)
//
// Round-3: LDS staging now uses CDNA5 GLOBAL_LOAD_ASYNC_TO_LDS_B128
// (ASYNCcnt-tracked, no VGPR round-trip) so the global->LDS DMA of chunk k+1
// truly overlaps the WMMA burst on chunk k. s_wait_asynccnt 0 + barrier per
// k-step; global_prefetch of the k+2 chunk keeps the async fetch in L2.

#define D_   1024
#define B_   8192
#define NBR_ 7
#define H_   16
#define HD_  64

typedef unsigned short u16;
typedef __attribute__((ext_vector_type(16))) __bf16 v16bf;
typedef __attribute__((ext_vector_type(8)))  float  v8f;
typedef __attribute__((ext_vector_type(8)))  u16    u16x8;
typedef __attribute__((ext_vector_type(16))) u16    u16x16;

__constant__ int c_src[NBR_][4] = {
    {0, 1, 0, 0}, {0, 2, 0, 0}, {0, 3, 0, 0},
    {0, 1, 2, 0}, {0, 1, 3, 0}, {0, 2, 3, 0}, {0, 1, 2, 3}};

// ---------- helpers ----------------------------------------------------------

__device__ __forceinline__ u16 f2bf(float x) {
  unsigned u = __builtin_bit_cast(unsigned, x);
  u += 0x7fffu + ((u >> 16) & 1u);          // round-to-nearest-even
  return (u16)(u >> 16);
}

__device__ __forceinline__ v8f vzero8() {
  v8f v;
#pragma unroll
  for (int j = 0; j < 8; ++j) v[j] = 0.0f;
  return v;
}

__device__ __forceinline__ v16bf splat_bf(float s) {
  __bf16 b = (__bf16)s;
  v16bf v;
#pragma unroll
  for (int j = 0; j < 16; ++j) v[j] = b;
  return v;
}

// Async DMA: 16 bytes per lane, global -> LDS, tracked by ASYNCcnt.
// LDS dest = low 32 bits of the generic shared pointer (gfx1250 maps
// addr[31:0] into the wave's LDS allocation).
__device__ __forceinline__ void async_b128(u16* lds_dst, const u16* gsrc) {
  unsigned loff = (unsigned)(uintptr_t)lds_dst;
  unsigned long long ga = (unsigned long long)(uintptr_t)gsrc;
  asm volatile("global_load_async_to_lds_b128 %0, %1, off"
               :: "v"(loff), "v"(ga) : "memory");
}

__device__ __forceinline__ void wait_async0() {
  asm volatile("s_wait_asynccnt 0" ::: "memory");
}

// Load one 16x32 bf16 A/B fragment from a row-major [rows][stride] u16 buffer.
// ISA 7.12.2 16-bit layout: lanes 0-15 hold K {0..7,16..23}, lanes 16-31 hold
// K {8..15,24..31} -> two contiguous 16B runs per lane.
__device__ __forceinline__ v16bf load_frag(const u16* base, int row, int stride,
                                           int lane) {
  const u16* p = base + (size_t)row * stride + ((lane >> 4) ? 8 : 0);
  u16x8 lo = *(const u16x8*)p;
  u16x8 hi = *(const u16x8*)(p + 16);
  u16x16 u;
#pragma unroll
  for (int j = 0; j < 8; ++j) { u[j] = lo[j]; u[j + 8] = hi[j]; }
  return __builtin_bit_cast(v16bf, u);
}

__device__ __forceinline__ v8f wmma_bf16(v16bf a, v16bf b, v8f c) {
  // v_wmma_f32_16x16x32_bf16: D = A(16x32) x B(32x16) + C
  return __builtin_amdgcn_wmma_f32_16x16x32_bf16(false, a, false, b, (short)0,
                                                 c, false, false);
}

// ---------- tiny precompute kernels -----------------------------------------

__global__ void cvt_kernel(const float* __restrict__ src, u16* __restrict__ dst,
                           int n) {
  int t = blockIdx.x * blockDim.x + threadIdx.x;
  if (t < n) dst[t] = f2bf(src[t]);
}

__global__ void qh_kernel(const float* __restrict__ tokens,
                          const float* __restrict__ w_in,
                          const float* __restrict__ b_in,
                          float* __restrict__ qh) {
  int i = blockIdx.y;
  int d = blockIdx.x * blockDim.x + threadIdx.x;
  const float* w = w_in + ((size_t)i * 3 * D_ + d) * D_;
  const float* t = tokens + (size_t)i * D_;
  float s = b_in[(size_t)i * 3 * D_ + d];
  for (int e = 0; e < D_; ++e) s = fmaf(t[e], w[e], s);
  qh[(size_t)i * D_ + d] = s;
}

__global__ void r_kernel(const float* __restrict__ w_in,
                         const float* __restrict__ qh, u16* __restrict__ Rb) {
  int i = blockIdx.z, h = blockIdx.y;
  int k = blockIdx.x * blockDim.x + threadIdx.x;
  const float* q = qh + (size_t)i * D_ + h * HD_;
  const float* w = w_in + ((size_t)i * 3 * D_ + D_ + h * HD_) * D_ + k;
  float s = 0.0f;
  for (int e = 0; e < HD_; ++e) s = fmaf(q[e], w[(size_t)e * D_], s);
  Rb[((size_t)i * H_ + h) * D_ + k] = f2bf(s * 0.125f);  // 0.125 = HD^-0.5
}

__global__ void cout_kernel(const float* __restrict__ w_out,
                            const float* __restrict__ b_in,
                            const float* __restrict__ b_out,
                            float* __restrict__ cout) {
  int i = blockIdx.y;
  int d = blockIdx.x * blockDim.x + threadIdx.x;
  const float* w = w_out + ((size_t)i * D_ + d) * D_;
  const float* bv = b_in + (size_t)i * 3 * D_ + 2 * D_;
  float s = b_out[(size_t)i * D_ + d];
  for (int e = 0; e < D_; ++e) s = fmaf(w[e], bv[e], s);
  cout[(size_t)i * D_ + d] = s;
}

// ---------- scores + softmax -------------------------------------------------
// One wave = 16 batch rows x 16 heads. acc[l] holds scores for kv entry l.
// Softmax over l is per-lane VALU: the l axis lives across accumulator regs.

template <int L>
__global__ void __launch_bounds__(256) scores_kernel(
    const u16* __restrict__ Xb, const u16* __restrict__ Rb,
    float* __restrict__ attn, int br0) {
  const int lane = threadIdx.x & 31;
  const int wave = threadIdx.x >> 5;
  const int br   = br0 + blockIdx.y;
  const int row0 = (blockIdx.x * 8 + wave) * 16;

  const u16* R = Rb + (size_t)br * H_ * D_;
  const u16* Xs[L];
#pragma unroll
  for (int l = 0; l < L; ++l)
    Xs[l] = Xb + (size_t)c_src[br][l] * ((size_t)B_ * D_);

  v8f acc[L];
#pragma unroll
  for (int l = 0; l < L; ++l) acc[l] = vzero8();

  for (int k0 = 0; k0 < D_; k0 += 32) {
    v16bf bf = load_frag(R + k0, lane & 15, D_, lane);
#pragma unroll
    for (int l = 0; l < L; ++l) {
      v16bf af = load_frag(Xs[l] + (size_t)row0 * D_ + k0, lane & 15, D_, lane);
      acc[l] = wmma_bf16(af, bf, acc[l]);
    }
  }

  const int h = lane & 15;
#pragma unroll
  for (int j = 0; j < 8; ++j) {
    int row = row0 + j + ((lane >> 4) << 3);
    float m = -3.0e38f;
#pragma unroll
    for (int l = 0; l < L; ++l) m = fmaxf(m, acc[l][j]);
    float e[L];
    float s = 0.0f;
#pragma unroll
    for (int l = 0; l < L; ++l) { e[l] = __expf(acc[l][j] - m); s += e[l]; }
    float inv = 1.0f / s;
#pragma unroll
    for (int l = 0; l < L; ++l)
      attn[(((size_t)(br * 4 + l)) * B_ + row) * H_ + h] = e[l] * inv;
  }
}

// ---------- O = (sum_l a_l (.) X_l) @ Wv^T ----------------------------------
// WG tile 128(M) x 256(N), 8 waves as 2M x 4N; each wave's 64 N-cols are one
// head, so a[l,row,head] folds into the A fragment (packed bf16 mul/add).
// Double-buffered LDS filled by async global->LDS DMA (overlaps WMMA burst).

template <int L>
__global__ void __launch_bounds__(256) ov_kernel(
    int br, const u16* __restrict__ Xb, const u16* __restrict__ Wv,
    const float* __restrict__ attn, u16* __restrict__ O) {
  __shared__ u16 sA[2][L][128 * 32];
  __shared__ u16 sB[2][256 * 32];

  const int tid = threadIdx.x, lane = tid & 31;
  const int wave = tid >> 5, wm = wave >> 2, wn = wave & 3;
  const int row0 = blockIdx.x * 128, col0 = blockIdx.y * 256;
  const int head = (col0 >> 6) + wn;

  const u16* Xs[L];
#pragma unroll
  for (int l = 0; l < L; ++l)
    Xs[l] = Xb + (size_t)c_src[br][l] * ((size_t)B_ * D_);

  float a_sc[4][L];
#pragma unroll
  for (int mf = 0; mf < 4; ++mf) {
    int r = row0 + wm * 64 + mf * 16 + (lane & 15);
#pragma unroll
    for (int l = 0; l < L; ++l)
      a_sc[mf][l] = attn[(((size_t)(br * 4 + l)) * B_ + r) * H_ + head];
  }

  v8f acc[4][4];
#pragma unroll
  for (int mf = 0; mf < 4; ++mf)
#pragma unroll
    for (int nf = 0; nf < 4; ++nf) acc[mf][nf] = vzero8();

  const int lr = tid >> 1;           // A staging: row
  const int lc = (tid & 1) * 16;     // A staging: col segment
  const u16* wsrc = Wv + ((size_t)br * D_ + col0 + tid) * D_;

  auto stage = [&](int k0, int buf) {
#pragma unroll
    for (int l = 0; l < L; ++l) {
      const u16* src = Xs[l] + (size_t)(row0 + lr) * D_ + k0 + lc;
      async_b128(&sA[buf][l][lr * 32 + lc], src);
      async_b128(&sA[buf][l][lr * 32 + lc + 8], src + 8);
      __builtin_prefetch((const void*)(src + 64), 0, 1);  // k+2 chunk -> L2
    }
    const u16* src = wsrc + k0;
    async_b128(&sB[buf][tid * 32], src);
    async_b128(&sB[buf][tid * 32 + 8], src + 8);
    async_b128(&sB[buf][tid * 32 + 16], src + 16);
    async_b128(&sB[buf][tid * 32 + 24], src + 24);
    __builtin_prefetch((const void*)(src + 64), 0, 1);
  };

  stage(0, 0);
  wait_async0();
  __syncthreads();

  int buf = 0;
  for (int k0 = 0; k0 < D_; k0 += 32) {
    if (k0 + 32 < D_) stage(k0 + 32, buf ^ 1);  // async DMA, fire-and-forget

    v16bf bf[4];
#pragma unroll
    for (int nf = 0; nf < 4; ++nf)
      bf[nf] = load_frag(sB[buf], wn * 64 + nf * 16 + (lane & 15), 32, lane);

#pragma unroll
    for (int mf = 0; mf < 4; ++mf) {
      int ar = wm * 64 + mf * 16 + (lane & 15);
      // A' = sum_l a_l (.) X_l  (packed bf16 mul/add)
      v16bf af = load_frag(sA[buf][0], ar, 32, lane) * splat_bf(a_sc[mf][0]);
#pragma unroll
      for (int l = 1; l < L; ++l)
        af = af + load_frag(sA[buf][l], ar, 32, lane) * splat_bf(a_sc[mf][l]);
#pragma unroll
      for (int nf = 0; nf < 4; ++nf)
        acc[mf][nf] = wmma_bf16(af, bf[nf], acc[mf][nf]);
    }

    wait_async0();      // my async stage of buf^1 landed in LDS
    __syncthreads();    // everyone's stage landed; reads of buf done
    buf ^= 1;
  }

#pragma unroll
  for (int mf = 0; mf < 4; ++mf)
#pragma unroll
    for (int nf = 0; nf < 4; ++nf)
#pragma unroll
      for (int j = 0; j < 8; ++j) {
        int r = row0 + wm * 64 + mf * 16 + j + ((lane >> 4) << 3);
        int c = col0 + wn * 64 + nf * 16 + (lane & 15);
        O[(size_t)r * D_ + c] = f2bf(acc[mf][nf][j]);
      }
}

// ---------- out = O @ Wo^T + c_out ------------------------------------------

__global__ void __launch_bounds__(256) proj_kernel(
    int br, const u16* __restrict__ O, const u16* __restrict__ Wo,
    const float* __restrict__ cout, float* __restrict__ out) {
  __shared__ u16 sA[2][128 * 32];
  __shared__ u16 sB[2][256 * 32];

  const int tid = threadIdx.x, lane = tid & 31;
  const int wave = tid >> 5, wm = wave >> 2, wn = wave & 3;
  const int row0 = blockIdx.x * 128, col0 = blockIdx.y * 256;

  v8f acc[4][4];
#pragma unroll
  for (int mf = 0; mf < 4; ++mf)
#pragma unroll
    for (int nf = 0; nf < 4; ++nf) acc[mf][nf] = vzero8();

  const int lr = tid >> 1;
  const int lc = (tid & 1) * 16;
  const u16* wsrc = Wo + ((size_t)br * D_ + col0 + tid) * D_;

  auto stage = [&](int k0, int buf) {
    {
      const u16* src = O + (size_t)(row0 + lr) * D_ + k0 + lc;
      async_b128(&sA[buf][lr * 32 + lc], src);
      async_b128(&sA[buf][lr * 32 + lc + 8], src + 8);
      __builtin_prefetch((const void*)(src + 64), 0, 1);
    }
    {
      const u16* src = wsrc + k0;
      async_b128(&sB[buf][tid * 32], src);
      async_b128(&sB[buf][tid * 32 + 8], src + 8);
      async_b128(&sB[buf][tid * 32 + 16], src + 16);
      async_b128(&sB[buf][tid * 32 + 24], src + 24);
      __builtin_prefetch((const void*)(src + 64), 0, 1);
    }
  };

  stage(0, 0);
  wait_async0();
  __syncthreads();

  int buf = 0;
  for (int k0 = 0; k0 < D_; k0 += 32) {
    if (k0 + 32 < D_) stage(k0 + 32, buf ^ 1);

    v16bf bf[4];
#pragma unroll
    for (int nf = 0; nf < 4; ++nf)
      bf[nf] = load_frag(sB[buf], wn * 64 + nf * 16 + (lane & 15), 32, lane);

#pragma unroll
    for (int mf = 0; mf < 4; ++mf) {
      v16bf af = load_frag(sA[buf], wm * 64 + mf * 16 + (lane & 15), 32, lane);
#pragma unroll
      for (int nf = 0; nf < 4; ++nf)
        acc[mf][nf] = wmma_bf16(af, bf[nf], acc[mf][nf]);
    }

    wait_async0();
    __syncthreads();
    buf ^= 1;
  }

#pragma unroll
  for (int mf = 0; mf < 4; ++mf)
#pragma unroll
    for (int nf = 0; nf < 4; ++nf)
#pragma unroll
      for (int j = 0; j < 8; ++j) {
        int r = row0 + wm * 64 + mf * 16 + j + ((lane >> 4) << 3);
        int c = col0 + wn * 64 + nf * 16 + (lane & 15);
        out[((size_t)br * B_ + r) * D_ + c] =
            acc[mf][nf][j] + cout[(size_t)br * D_ + c];
      }
}

// ---------- host orchestration ----------------------------------------------

extern "C" void kernel_launch(void* const* d_in, const int* in_sizes, int n_in,
                              void* d_out, int out_size, void* d_ws,
                              size_t ws_size, hipStream_t stream) {
  (void)in_sizes; (void)n_in; (void)out_size; (void)ws_size;
  const float* xsrc[4] = {(const float*)d_in[0], (const float*)d_in[1],
                          (const float*)d_in[2], (const float*)d_in[3]};
  const float* tokens = (const float*)d_in[4];
  const float* w_in   = (const float*)d_in[5];
  const float* b_in   = (const float*)d_in[6];
  const float* w_out  = (const float*)d_in[7];
  const float* b_out  = (const float*)d_in[8];
  float* out = (float*)d_out;

  char* ws = (char*)d_ws;
  size_t off = 0;
  auto take = [&](size_t bytes) -> char* {
    char* p = ws + off;
    off += (bytes + 255) & ~(size_t)255;
    return p;
  };
  float* qh   = (float*)take((size_t)NBR_ * D_ * sizeof(float));
  float* cout = (float*)take((size_t)NBR_ * D_ * sizeof(float));
  u16*   Rb   = (u16*)  take((size_t)NBR_ * H_ * D_ * sizeof(u16));
  float* attn = (float*)take((size_t)NBR_ * 4 * B_ * H_ * sizeof(float));
  u16*   Xb   = (u16*)  take((size_t)4 * B_ * D_ * sizeof(u16));
  u16*   Wv   = (u16*)  take((size_t)NBR_ * D_ * D_ * sizeof(u16));
  u16*   Wo   = (u16*)  take((size_t)NBR_ * D_ * D_ * sizeof(u16));
  u16*   O    = (u16*)  take((size_t)B_ * D_ * sizeof(u16));

  // bf16 packing
  for (int s = 0; s < 4; ++s)
    cvt_kernel<<<(B_ * D_) / 256, 256, 0, stream>>>(
        xsrc[s], Xb + (size_t)s * B_ * D_, B_ * D_);
  for (int i = 0; i < NBR_; ++i)
    cvt_kernel<<<(D_ * D_) / 256, 256, 0, stream>>>(
        w_in + ((size_t)i * 3 * D_ + 2 * D_) * D_, Wv + (size_t)i * D_ * D_,
        D_ * D_);
  cvt_kernel<<<(NBR_ * D_ * D_) / 256, 256, 0, stream>>>(w_out, Wo,
                                                         NBR_ * D_ * D_);

  // tiny precompute
  qh_kernel<<<dim3(D_ / 256, NBR_), 256, 0, stream>>>(tokens, w_in, b_in, qh);
  r_kernel<<<dim3(D_ / 256, H_, NBR_), 256, 0, stream>>>(w_in, qh, Rb);
  cout_kernel<<<dim3(D_ / 256, NBR_), 256, 0, stream>>>(w_out, b_in, b_out,
                                                        cout);

  // scores + softmax, grouped by L so the kernels specialize at compile time
  scores_kernel<2><<<dim3(B_ / 128, 3), 256, 0, stream>>>(Xb, Rb, attn, 0);
  scores_kernel<3><<<dim3(B_ / 128, 3), 256, 0, stream>>>(Xb, Rb, attn, 3);
  scores_kernel<4><<<dim3(B_ / 128, 1), 256, 0, stream>>>(Xb, Rb, attn, 6);

  // per-branch: attention-mixed V GEMM, then output projection
  static const int hL[NBR_] = {2, 2, 2, 3, 3, 3, 4};
  for (int br = 0; br < NBR_; ++br) {
    dim3 g(B_ / 128, D_ / 256);
    switch (hL[br]) {
      case 2: ov_kernel<2><<<g, 256, 0, stream>>>(br, Xb, Wv, attn, O); break;
      case 3: ov_kernel<3><<<g, 256, 0, stream>>>(br, Xb, Wv, attn, O); break;
      default: ov_kernel<4><<<g, 256, 0, stream>>>(br, Xb, Wv, attn, O); break;
    }
    proj_kernel<<<g, 256, 0, stream>>>(br, O, Wo, cout, out);
  }
}
